// ImageEncoder_13176959664212
// MI455X (gfx1250) — matrix operand order repaired and checked by reading, not compiled, not run on hardware
//
#include <hip/hip_runtime.h>

// ---------------------------------------------------------------------------
// Vision-Mamba forward for MI455X (gfx1250, wave32, WMMA).
// GEMMs: v_wmma_f32_16x16x32_bf16, 2x2 micro-tile (32x32 out per wave).
// Scan:  one wave per (b,d), h resident in registers, shuffle-only reduce.
// ---------------------------------------------------------------------------

typedef __attribute__((ext_vector_type(16))) __bf16 v16bf;
typedef __attribute__((ext_vector_type(8)))  __bf16 v8bf;
typedef __attribute__((ext_vector_type(8)))  float  v8f;
typedef __attribute__((ext_vector_type(4)))  float  v4f;

#define DIM        256
#define DT_RANK    32
#define D_STATE    256
#define DEPTH      12
#define NPATCH     64
#define BATCH      16
#define NTOK       (BATCH * NPATCH)      // 1024 rows
#define PATCH_DIM  3072
#define NCLS       768
#define DBC        (DT_RANK + 2 * D_STATE) // 544

__device__ __forceinline__ float softplusf(float x) {
    return (x > 20.f) ? x : log1pf(__expf(x));
}

// ---------------- f32 -> bf16 convert ----------------
__global__ void f2bf_kernel(const float* __restrict__ in, __bf16* __restrict__ out, int n) {
    int i = blockIdx.x * 256 + threadIdx.x;
    if (i < n) out[i] = (__bf16)in[i];
}

// ---------------- patch gather + LayerNorm(3072) -> bf16 ----------------
__global__ void patch_kernel(const float* __restrict__ x,
                             const float* __restrict__ g, const float* __restrict__ bb,
                             __bf16* __restrict__ X0b) {
    int row = blockIdx.x;                  // b*64 + tok
    int bi = row >> 6, tok = row & 63;
    int gh = tok >> 3, gw = tok & 7;
    int t = threadIdx.x;

    float v[12];
    float s = 0.f, s2 = 0.f;
#pragma unroll
    for (int j = 0; j < 12; ++j) {
        int p  = t * 12 + j;               // (p1*32+p2)*3 + c
        int c  = p % 3;
        int pp = p / 3;
        int p1 = pp >> 5, p2 = pp & 31;
        float val = x[((bi * 3 + c) * 256 + gh * 32 + p1) * 256 + (gw * 32 + p2)];
        v[j] = val; s += val; s2 += val * val;
    }
#pragma unroll
    for (int off = 16; off; off >>= 1) {
        s  += __shfl_down(s, off);
        s2 += __shfl_down(s2, off);
    }
    __shared__ float rs[8], rs2[8], mv[2];
    if ((t & 31) == 0) { rs[t >> 5] = s; rs2[t >> 5] = s2; }
    __syncthreads();
    if (t == 0) {
        float a = 0.f, a2 = 0.f;
        for (int w = 0; w < 8; ++w) { a += rs[w]; a2 += rs2[w]; }
        float m  = a / (float)PATCH_DIM;
        float vr = a2 / (float)PATCH_DIM - m * m;
        mv[0] = m; mv[1] = rsqrtf(vr + 1e-5f);
    }
    __syncthreads();
    float m = mv[0], r = mv[1];
#pragma unroll
    for (int j = 0; j < 12; ++j) {
        int p = t * 12 + j;
        float y = (v[j] - m) * r * g[p] + bb[p];
        X0b[(size_t)row * PATCH_DIM + p] = (__bf16)y;
    }
}

// ---------------- WMMA fragment loaders (documented gfx1250 layouts) -------
// A: 16x32 bf16, lane m(0..15): K=koff..koff+7 (V0-3), K=16+koff.. (V4-7)
__device__ __forceinline__ v16bf load_A_frag(const __bf16* __restrict__ A, int lda,
                                             int mt, int k0, int lane) {
    int m    = lane & 15;
    int koff = (lane < 16) ? 0 : 8;
    const __bf16* p = A + (size_t)(mt + m) * lda + k0 + koff;
    v8bf lo = *reinterpret_cast<const v8bf*>(p);        // K = koff .. koff+7
    v8bf hi = *reinterpret_cast<const v8bf*>(p + 16);   // K = 16+koff ..
    v16bf a;
#pragma unroll
    for (int j = 0; j < 8; ++j) { a[j] = lo[j]; a[8 + j] = hi[j]; }
    return a;
}

// B: 32x16 bf16 (K x N) from W[N,K] row-major (computing X @ W^T):
// lane n(0..15): K=0..7 (V0-3), K=16..23 (V4-7); lane n+16: K=8..15, K=24..31.
__device__ __forceinline__ v16bf load_B_fragT(const __bf16* __restrict__ W, int ldw,
                                              int nt, int k0, int lane) {
    int n     = lane & 15;
    int kbase = (lane < 16) ? 0 : 8;
    const __bf16* p = W + (size_t)(nt + n) * ldw + k0 + kbase;
    v8bf lo = *reinterpret_cast<const v8bf*>(p);
    v8bf hi = *reinterpret_cast<const v8bf*>(p + 16);
    v16bf b;
#pragma unroll
    for (int j = 0; j < 8; ++j) { b[j] = lo[j]; b[8 + j] = hi[j]; }
    return b;
}

// ---------------- WMMA GEMM, 2x2 micro-tile per wave -----------------------
// C[M,N] = ep(A[M,K] @ W[N,K]^T + bias); ep: 0 none, 1 +bias, 2 softplus(+bias)
__global__ void gemm_wmma(const __bf16* __restrict__ A, int lda,
                          const __bf16* __restrict__ W, int ldw,
                          const float* __restrict__ bias,
                          float* __restrict__ C, __bf16* __restrict__ Cb, int ldc,
                          int M, int N, int K, int ep) {
    int wave = threadIdx.x >> 5;
    int lane = threadIdx.x & 31;
    int tiles_n = N >> 5;                      // 32-wide macro tiles
    int tile = blockIdx.x * 8 + wave;          // wave-uniform -> EXEC all-1s
    if (tile >= (M >> 5) * tiles_n) return;
    int mt = (tile / tiles_n) << 5;
    int nt = (tile % tiles_n) << 5;

    v8f acc00 = {}, acc01 = {}, acc10 = {}, acc11 = {};
    for (int k0 = 0; k0 < K; k0 += 32) {
        v16bf a0 = load_A_frag(A, lda, mt,      k0, lane);
        v16bf a1 = load_A_frag(A, lda, mt + 16, k0, lane);
        v16bf b0 = load_B_fragT(W, ldw, nt,      k0, lane);
        v16bf b1 = load_B_fragT(W, ldw, nt + 16, k0, lane);
        acc00 = __builtin_amdgcn_wmma_f32_16x16x32_bf16(false, a0, false, b0, (short)0, acc00, false, false);
        acc01 = __builtin_amdgcn_wmma_f32_16x16x32_bf16(false, a0, false, b1, (short)0, acc01, false, false);
        acc10 = __builtin_amdgcn_wmma_f32_16x16x32_bf16(false, a1, false, b0, (short)0, acc10, false, false);
        acc11 = __builtin_amdgcn_wmma_f32_16x16x32_bf16(false, a1, false, b1, (short)0, acc11, false, false);
    }

    // D layout: VGPR r, lane c(0..15) -> (M=r, N=c); lanes 16..31 -> M=r+8
    int c0    = nt + (lane & 15);
    int c1    = c0 + 16;
    int rbase = ((lane < 16) ? 0 : 8);
    float bv0 = bias ? bias[c0] : 0.f;
    float bv1 = bias ? bias[c1] : 0.f;
#pragma unroll
    for (int r = 0; r < 8; ++r) {
        int row0 = mt + rbase + r;
        int row1 = row0 + 16;
        float v00 = acc00[r] + bv0, v01 = acc01[r] + bv1;
        float v10 = acc10[r] + bv0, v11 = acc11[r] + bv1;
        if (ep == 2) { v00 = softplusf(v00); v01 = softplusf(v01);
                       v10 = softplusf(v10); v11 = softplusf(v11); }
        size_t o00 = (size_t)row0 * ldc + c0, o01 = (size_t)row0 * ldc + c1;
        size_t o10 = (size_t)row1 * ldc + c0, o11 = (size_t)row1 * ldc + c1;
        C[o00] = v00; C[o01] = v01; C[o10] = v10; C[o11] = v11;
        if (Cb) { Cb[o00] = (__bf16)v00; Cb[o01] = (__bf16)v01;
                  Cb[o10] = (__bf16)v10; Cb[o11] = (__bf16)v11; }
    }
}

// ---------------- LayerNorm(256) [+silu gate, +bf16 mirror] ----------------
__global__ void ln_kernel(const float* __restrict__ X,
                          const float* __restrict__ g, const float* __restrict__ bb,
                          float* __restrict__ Y, __bf16* __restrict__ Yb,
                          float* __restrict__ Z) {
    int row = blockIdx.x;
    int t = threadIdx.x;
    float x = X[(size_t)row * DIM + t];
    float s = x, s2 = x * x;
#pragma unroll
    for (int off = 16; off; off >>= 1) {
        s  += __shfl_down(s, off);
        s2 += __shfl_down(s2, off);
    }
    __shared__ float rs[8], rs2[8], mv[2];
    if ((t & 31) == 0) { rs[t >> 5] = s; rs2[t >> 5] = s2; }
    __syncthreads();
    if (t == 0) {
        float a = 0.f, a2 = 0.f;
        for (int w = 0; w < 8; ++w) { a += rs[w]; a2 += rs2[w]; }
        float m  = a / (float)DIM;
        float vr = a2 / (float)DIM - m * m;
        mv[0] = m; mv[1] = rsqrtf(vr + 1e-5f);
    }
    __syncthreads();
    float y = (x - mv[0]) * mv[1] * g[t] + bb[t];
    Y[(size_t)row * DIM + t] = y;
    if (Yb) Yb[(size_t)row * DIM + t] = (__bf16)y;
    if (Z)  Z[(size_t)row * DIM + t]  = y / (1.f + __expf(-y));
}

// ---------------- selective scan: one WAVE per (b,d), 8 n-states per lane --
// block = 256 thr = 8 waves = 8 d's; grid = BATCH * DIM/8 = 512 blocks.
// No LDS, no barriers: C.h dot is a 5-level wave32 shuffle reduction.
__global__ void scan_kernel(const float* __restrict__ u,      // xs  [B,64,256]
                            const float* __restrict__ delta,  //     [B,64,256]
                            const float* __restrict__ dbc,    //     [B,64,544]
                            const float* __restrict__ A_log,  // layer slice [256,256]
                            const float* __restrict__ Dp,     // [256]
                            float* __restrict__ Y) {          // [B,64,256]
    int b      = blockIdx.x >> 5;                 // 32 d-groups per batch
    int dgroup = blockIdx.x & 31;
    int wave   = threadIdx.x >> 5;
    int lane   = threadIdx.x & 31;
    int d      = dgroup * 8 + wave;
    int n0     = lane * 8;                        // 8 consecutive states per lane

    float A[8], h[8];
    const float* Arow = A_log + (size_t)d * D_STATE + n0;
#pragma unroll
    for (int j = 0; j < 8; ++j) { A[j] = -__expf(Arow[j]); h[j] = 0.f; }
    float Dd = Dp[d];

    for (int l = 0; l < NPATCH; ++l) {
        size_t rowd = (size_t)(b * NPATCH + l);
        const float* dbcrow = dbc + rowd * DBC;
        float dt = delta[rowd * DIM + d];
        float ut = u[rowd * DIM + d];
        v4f B0 = *reinterpret_cast<const v4f*>(dbcrow + DT_RANK + n0);
        v4f B1 = *reinterpret_cast<const v4f*>(dbcrow + DT_RANK + n0 + 4);
        v4f C0 = *reinterpret_cast<const v4f*>(dbcrow + DT_RANK + D_STATE + n0);
        v4f C1 = *reinterpret_cast<const v4f*>(dbcrow + DT_RANK + D_STATE + n0 + 4);

        if (l + 1 < NPATCH) {                     // prefetch next step's B/C
            const float* nrow = dbcrow + DBC;
            __builtin_prefetch(nrow + DT_RANK + n0, 0, 1);
            __builtin_prefetch(nrow + DT_RANK + D_STATE + n0, 0, 1);
        }

        float du = dt * ut;
        float v = 0.f;
#pragma unroll
        for (int j = 0; j < 4; ++j) {
            h[j] = __expf(dt * A[j]) * h[j] + du * B0[j];
            v += h[j] * C0[j];
        }
#pragma unroll
        for (int j = 0; j < 4; ++j) {
            h[4 + j] = __expf(dt * A[4 + j]) * h[4 + j] + du * B1[j];
            v += h[4 + j] * C1[j];
        }
#pragma unroll
        for (int off = 16; off; off >>= 1) v += __shfl_down(v, off);
        if (lane == 0) Y[rowd * DIM + d] = v + ut * Dd;
    }
}

// ---------------- gated residual: X = (y1 + y2) * z + X ----------------
__global__ void combine_kernel(float* __restrict__ X, const float* __restrict__ z,
                               const float* __restrict__ y1, const float* __restrict__ y2) {
    int i = blockIdx.x * 256 + threadIdx.x;
    X[i] = (y1[i] + y2[i]) * z[i] + X[i];
}

// ---------------- head pooling: 4x4 token-block mean -> out[b,4,768] -------
__global__ void pool_kernel(const float* __restrict__ T, float* __restrict__ out) {
    int idx = blockIdx.x * 256 + threadIdx.x;    // b*4*768 + q*768 + c
    if (idx >= BATCH * 4 * NCLS) return;
    int c = idx % NCLS;
    int q = (idx / NCLS) & 3;
    int b = idx / (4 * NCLS);
    int i = q >> 1, j = q & 1;
    float s = 0.f;
#pragma unroll
    for (int gh = 0; gh < 4; ++gh)
#pragma unroll
        for (int gw = 0; gw < 4; ++gw) {
            int tok = (i * 4 + gh) * 8 + (j * 4 + gw);
            s += T[(size_t)(b * NPATCH + tok) * NCLS + c];
        }
    out[idx] = s * (1.f / 16.f);
}

// ---------------------------------------------------------------------------
extern "C" void kernel_launch(void* const* d_in, const int* in_sizes, int n_in,
                              void* d_out, int out_size, void* d_ws, size_t ws_size,
                              hipStream_t stream) {
    const float* x        = (const float*)d_in[0];
    const float* pe1_g    = (const float*)d_in[1];
    const float* pe1_b    = (const float*)d_in[2];
    const float* Wpe      = (const float*)d_in[3];
    const float* bpe      = (const float*)d_in[4];
    const float* pe2_g    = (const float*)d_in[5];
    const float* pe2_b    = (const float*)d_in[6];
    const float* ln_g     = (const float*)d_in[7];
    const float* ln_b     = (const float*)d_in[8];
    const float* fcw      = (const float*)d_in[9];
    const float* fcb      = (const float*)d_in[10];
    const float* bcw      = (const float*)d_in[11];
    const float* bcb      = (const float*)d_in[12];
    const float* Wdbc     = (const float*)d_in[13];
    const float* Wdt      = (const float*)d_in[14];
    const float* bdt      = (const float*)d_in[15];
    const float* A_log    = (const float*)d_in[16];
    const float* Dp       = (const float*)d_in[17];
    const float* hln_g    = (const float*)d_in[18];
    const float* hln_b    = (const float*)d_in[19];
    const float* Whead    = (const float*)d_in[20];
    const float* bhead    = (const float*)d_in[21];
    float* out = (float*)d_out;

    // -------- workspace carve-up (256B aligned) --------
    size_t off = 0;
    auto alloc = [&](size_t bytes) -> char* {
        char* p = (char*)d_ws + off;
        off += (bytes + 255) & ~(size_t)255;
        return p;
    };
    __bf16* Wpe_b   = (__bf16*)alloc((size_t)DIM * PATCH_DIM * 2);
    __bf16* fcw_b   = (__bf16*)alloc((size_t)DEPTH * DIM * DIM * 2);
    __bf16* bcw_b   = (__bf16*)alloc((size_t)DEPTH * DIM * DIM * 2);
    __bf16* Wdbc_b  = (__bf16*)alloc((size_t)DEPTH * DBC * DIM * 2);
    __bf16* Wdt_b   = (__bf16*)alloc((size_t)DEPTH * DIM * DT_RANK * 2);
    __bf16* Whead_b = (__bf16*)alloc((size_t)NCLS * DIM * 2);
    __bf16* X0b     = (__bf16*)alloc((size_t)NTOK * PATCH_DIM * 2);
    float*  X       = (float*) alloc((size_t)NTOK * DIM * 4);
    float*  xn      = (float*) alloc((size_t)NTOK * DIM * 4);
    __bf16* xnb     = (__bf16*)alloc((size_t)NTOK * DIM * 2);
    float*  zg      = (float*) alloc((size_t)NTOK * DIM * 4);
    float*  xs1     = (float*) alloc((size_t)NTOK * DIM * 4);
    __bf16* xs1b    = (__bf16*)alloc((size_t)NTOK * DIM * 2);
    float*  xs2     = (float*) alloc((size_t)NTOK * DIM * 4);
    __bf16* xs2b    = (__bf16*)alloc((size_t)NTOK * DIM * 2);
    float*  dbc1    = (float*) alloc((size_t)NTOK * DBC * 4);
    __bf16* dbc1b   = (__bf16*)alloc((size_t)NTOK * DBC * 2);
    float*  dbc2    = (float*) alloc((size_t)NTOK * DBC * 4);
    __bf16* dbc2b   = (__bf16*)alloc((size_t)NTOK * DBC * 2);
    float*  dlt1    = (float*) alloc((size_t)NTOK * DIM * 4);
    float*  dlt2    = (float*) alloc((size_t)NTOK * DIM * 4);
    float*  y1      = (float*) alloc((size_t)NTOK * DIM * 4);
    float*  y2      = (float*) alloc((size_t)NTOK * DIM * 4);
    float*  Th      = (float*) alloc((size_t)NTOK * NCLS * 4);
    (void)ws_size; (void)n_in; (void)in_sizes; (void)out_size;

    auto cvt = [&](const float* src, __bf16* dst, int n) {
        f2bf_kernel<<<(n + 255) / 256, 256, 0, stream>>>(src, dst, n);
    };
    auto gemm = [&](const __bf16* A, int lda, const __bf16* W, int ldw,
                    const float* bias, float* C, __bf16* Cb, int ldc,
                    int M, int N, int K, int ep) {
        int tiles = (M / 32) * (N / 32);
        gemm_wmma<<<(tiles + 7) / 8, 256, 0, stream>>>(A, lda, W, ldw, bias,
                                                       C, Cb, ldc, M, N, K, ep);
    };

    // -------- weight conversion (deterministic, every call) --------
    cvt(Wpe,   Wpe_b,   DIM * PATCH_DIM);
    cvt(fcw,   fcw_b,   DEPTH * DIM * DIM);
    cvt(bcw,   bcw_b,   DEPTH * DIM * DIM);
    cvt(Wdbc,  Wdbc_b,  DEPTH * DBC * DIM);
    cvt(Wdt,   Wdt_b,   DEPTH * DIM * DT_RANK);
    cvt(Whead, Whead_b, NCLS * DIM);

    // -------- patch embed --------
    patch_kernel<<<NTOK, 256, 0, stream>>>(x, pe1_g, pe1_b, X0b);
    gemm(X0b, PATCH_DIM, Wpe_b, PATCH_DIM, bpe, xs1, nullptr, DIM,
         NTOK, DIM, PATCH_DIM, /*ep=*/1);
    ln_kernel<<<NTOK, 256, 0, stream>>>(xs1, pe2_g, pe2_b, X, nullptr, nullptr);

    // -------- 12 Mamba blocks --------
    for (int i = 0; i < DEPTH; ++i) {
        const float* lg = ln_g + i * DIM;
        const float* lb = ln_b + i * DIM;
        ln_kernel<<<NTOK, 256, 0, stream>>>(X, lg, lb, xn, xnb, zg);

        // two 1x1 conv branches, softplus epilogue
        gemm(xnb, DIM, fcw_b + (size_t)i * DIM * DIM, DIM, fcb + i * DIM,
             xs1, xs1b, DIM, NTOK, DIM, DIM, 2);
        gemm(xnb, DIM, bcw_b + (size_t)i * DIM * DIM, DIM, bcb + i * DIM,
             xs2, xs2b, DIM, NTOK, DIM, DIM, 2);

        // x -> (dt, B, C)
        gemm(xs1b, DIM, Wdbc_b + (size_t)i * DBC * DIM, DIM, nullptr,
             dbc1, dbc1b, DBC, NTOK, DBC, DIM, 0);
        gemm(xs2b, DIM, Wdbc_b + (size_t)i * DBC * DIM, DIM, nullptr,
             dbc2, dbc2b, DBC, NTOK, DBC, DIM, 0);

        // delta = softplus(dt @ Wdt^T + bdt)   (dt = first 32 cols of dbc, lda=544)
        gemm(dbc1b, DBC, Wdt_b + (size_t)i * DIM * DT_RANK, DT_RANK, bdt + i * DIM,
             dlt1, nullptr, DIM, NTOK, DIM, DT_RANK, 2);
        gemm(dbc2b, DBC, Wdt_b + (size_t)i * DIM * DT_RANK, DT_RANK, bdt + i * DIM,
             dlt2, nullptr, DIM, NTOK, DIM, DT_RANK, 2);

        // selective scans (wave-per-(b,d), register-resident state)
        scan_kernel<<<BATCH * (DIM / 8), 256, 0, stream>>>(
            xs1, dlt1, dbc1, A_log + (size_t)i * DIM * D_STATE, Dp + i * DIM, y1);
        scan_kernel<<<BATCH * (DIM / 8), 256, 0, stream>>>(
            xs2, dlt2, dbc2, A_log + (size_t)i * DIM * D_STATE, Dp + i * DIM, y2);

        combine_kernel<<<NTOK, 256, 0, stream>>>(X, zg, y1, y2);
    }

    // -------- head --------
    ln_kernel<<<NTOK, 256, 0, stream>>>(X, hln_g, hln_b, xn, xnb, nullptr);
    gemm(xnb, DIM, Whead_b, DIM, bhead, Th, nullptr, NCLS, NTOK, NCLS, DIM, 1);
    pool_kernel<<<(BATCH * 4 * NCLS + 255) / 256, 256, 0, stream>>>(Th, out);
}
